// EGNN_20048907337763
// MI455X (gfx1250) — matrix operand order repaired and checked by reading, not compiled, not run on hardware
//
#include <hip/hip_runtime.h>
#include <stdint.h>

#define HD 256

typedef __attribute__((ext_vector_type(16))) _Float16 v16h;
typedef __attribute__((ext_vector_type(8)))  float    v8f;

union FragH { v16h h; uint4 q[2]; };

__device__ __forceinline__ float siluf(float v) { return v / (1.0f + __expf(-v)); }

__device__ __forceinline__ void atomicAddF(float* p, float v) {
  __hip_atomic_fetch_add(p, v, __ATOMIC_RELAXED, __HIP_MEMORY_SCOPE_AGENT);
}

__device__ __forceinline__ v8f vzero8() {
  v8f z;
  #pragma unroll
  for (int i = 0; i < 8; ++i) z[i] = 0.f;
  return z;
}

// One wave computes a 2x2 grid of 16x16 output tiles:
//   D[mt][nt] for mt in {0,1} (rows 0..15 / 16..31) and nt in {nt0, nt1}.
// A lives in LDS (f16 row-major, row stride = STRIDE halves, 32 rows);
// B lives in global, pre-swizzled to WMMA fragment order:
//   word[((kb*16+nt)*32+lane)*8+p] = pack(f16 W[k0][n], f16 W[k0+1][n]),
//   n=(lane&15)+nt*16, k0=kb*32+(lane>>4)*16+2p.
// Per K-step: 2 A-frags + 2 B-frags -> 4 WMMAs (2x reuse on both paths).
template<int KB, int STRIDE>
__device__ __forceinline__ void gemm_quad(const _Float16* aBase,
                                          const uint32_t* __restrict__ B,
                                          int nt0, int nt1, int lane,
                                          v8f& a00, v8f& a01, v8f& a10, v8f& a11)
{
  const int m  = lane & 15;
  const int hi = lane >> 4;
  const _Float16* arow0 = aBase + m * STRIDE + hi * 16;
  const _Float16* arow1 = aBase + (m + 16) * STRIDE + hi * 16;
  #pragma unroll
  for (int kb = 0; kb < KB; ++kb) {
    FragH A0, A1, B0, B1;
    // A fragment in contiguous K order (matches B packing): lanes<16 K={0..15}, lanes>=16 K={16..31}
    const uint4* ap0 = (const uint4*)(arow0 + kb * 32);
    A0.q[0] = ap0[0]; A0.q[1] = ap0[1];
    const uint4* ap1 = (const uint4*)(arow1 + kb * 32);
    A1.q[0] = ap1[0]; A1.q[1] = ap1[1];
    const uint4* bp0 = (const uint4*)(B + (size_t)((kb * 16 + nt0) * 32 + lane) * 8);
    B0.q[0] = bp0[0]; B0.q[1] = bp0[1];
    const uint4* bp1 = (const uint4*)(B + (size_t)((kb * 16 + nt1) * 32 + lane) * 8);
    B1.q[0] = bp1[0]; B1.q[1] = bp1[1];
    a00 = __builtin_amdgcn_wmma_f32_16x16x32_f16(false, A0.h, false, B0.h, (short)0, a00, false, false);
    a01 = __builtin_amdgcn_wmma_f32_16x16x32_f16(false, A0.h, false, B1.h, (short)0, a01, false, false);
    a10 = __builtin_amdgcn_wmma_f32_16x16x32_f16(false, A1.h, false, B0.h, (short)0, a10, false, false);
    a11 = __builtin_amdgcn_wmma_f32_16x16x32_f16(false, A1.h, false, B1.h, (short)0, a11, false, false);
  }
}

// SiLU epilogue for a 2x2 tile -> f16 LDS buffer with row stride 256.
__device__ __forceinline__ void silu_store4(const v8f& a00, const v8f& a01,
                                            const v8f& a10, const v8f& a11,
                                            float b0, float b1, int hi,
                                            int n0, int n1, _Float16* dst)
{
  #pragma unroll
  for (int r = 0; r < 8; ++r) {
    const int row = r + 8 * hi;          // D layout: VGPR r -> M=r (+8 for hi lanes)
    dst[row * 256 + n0]        = (_Float16)siluf(a00[r] + b0);
    dst[row * 256 + n1]        = (_Float16)siluf(a01[r] + b1);
    dst[(row + 16) * 256 + n0] = (_Float16)siluf(a10[r] + b0);
    dst[(row + 16) * 256 + n1] = (_Float16)siluf(a11[r] + b1);
  }
}

// ---------------------------------------------------------------------------
// Weight conversion: f32 [Ksrc,256] -> f16 WMMA-B fragment-ordered words.
// permute=1 applies the We1 row remap: [h_i(512) | dist(1) | eattr(32)] ->
// [h_i(512) | eattr(32) | dist(1) | zero-pad].
__global__ void __launch_bounds__(256)
convert_weights(const float* __restrict__ W, uint32_t* __restrict__ out,
                int KB, int Ksrc, int srcStride, int dstStride, int permute)
{
  const int words = KB * 4096;            // KB * 16 nt * 32 lanes * 8 words
  const int t = blockIdx.x * 256 + threadIdx.x;
  if (t >= words) return;
  const int l    = blockIdx.y;
  const int pp   = t & 7;
  const int lane = (t >> 3) & 31;
  const int nt   = (t >> 8) & 15;
  const int kb   = t >> 12;
  const int n    = (lane & 15) + nt * 16;
  const int hi   = lane >> 4;
  const int k0   = kb * 32 + hi * 16 + 2 * pp;

  float v[2];
  #pragma unroll
  for (int s = 0; s < 2; ++s) {
    int k = k0 + s;
    int ks = k;
    if (permute) {
      if (k < 512)       ks = k;
      else if (k < 544)  ks = k + 1;   // edge_attr rows (src 513..544)
      else if (k == 544) ks = 512;     // dist_sq row
      else               ks = -1;      // zero pad
    }
    float val = 0.f;
    if (ks >= 0 && ks < Ksrc) val = W[(size_t)l * srcStride + (size_t)ks * 256 + n];
    v[s] = val;
  }
  union { uint32_t u; _Float16 f[2]; } pk;
  pk.f[0] = (_Float16)v[0];
  pk.f[1] = (_Float16)v[1];
  out[(size_t)l * dstStride + t] = pk.u;
}

__global__ void __launch_bounds__(256)
zero_kernel(float* __restrict__ p, int n)
{
  int i = blockIdx.x * 256 + threadIdx.x;
  if (i < n) p[i] = 0.f;
}

// f32 -> f16 shadow copy of h
__global__ void __launch_bounds__(256)
cvt_h16_kernel(const float* __restrict__ src, _Float16* __restrict__ dst, int n)
{
  int i = blockIdx.x * 256 + threadIdx.x;
  if (i < n) dst[i] = (_Float16)src[i];
}

// ---------------------------------------------------------------------------
// Edge kernel: per block, 32 edges. Fused msg-build -> GEMM1 -> GEMM2 ->
// coord MLP -> atomic aggregation into m_agg / x_agg.
// h is gathered from the persistent f16 shadow (half the L2 gather bytes,
// no per-edge f32->f16 conversion VALU).
__global__ void __launch_bounds__(256, 2)
edge_kernel(const _Float16* __restrict__ h16, const float* __restrict__ x,
            const float* __restrict__ edge_attr, const int* __restrict__ eidx, int E,
            const uint32_t* __restrict__ We1s, const float* __restrict__ be1,
            const uint32_t* __restrict__ We2s, const float* __restrict__ be2,
            const uint32_t* __restrict__ Wx1s, const float* __restrict__ bx1,
            const float* __restrict__ Wx2,
            float* __restrict__ m_agg, float* __restrict__ x_agg)
{
  __shared__ alignas(16) _Float16 sMsg[32][576];   // msg tile, later reused for m
  __shared__ alignas(16) _Float16 sBuf[32][256];   // m1 tile, later t tile
  __shared__ float sDiff[32][3];
  __shared__ float sRed[32][16];
  __shared__ int sI[32], sJ[32];

  const int tid = threadIdx.x;
  const int eg0 = blockIdx.x * 32;

  if (tid < 32) {
    const int e = eg0 + tid;
    const int j = eidx[e];        // source  (edge_index[0])
    const int i = eidx[E + e];    // target  (edge_index[1])
    sI[tid] = i; sJ[tid] = j;
    float ds = 0.f;
    #pragma unroll
    for (int d = 0; d < 3; ++d) {
      float df = x[i * 3 + d] - x[j * 3 + d];
      sDiff[tid][d] = df;
      ds += df * df;
    }
    sMsg[tid][544] = (_Float16)ds;
    for (int k = 545; k < 576; ++k) sMsg[tid][k] = (_Float16)0.f;
  }
  __syncthreads();

  {
    const int e  = tid >> 3;            // 0..31
    const int c0 = (tid & 7) * 32;      // 8 chunks of 32 columns
    const int ni = sI[e], nj = sJ[e];
    // f16 gather: straight 16B copies, no conversion
    const uint4* hi4 = (const uint4*)(h16 + (size_t)ni * HD + c0);
    const uint4* hj4 = (const uint4*)(h16 + (size_t)nj * HD + c0);
    uint4* di = (uint4*)(&sMsg[e][c0]);
    uint4* dj = (uint4*)(&sMsg[e][256 + c0]);
    #pragma unroll
    for (int q = 0; q < 4; ++q) { di[q] = hi4[q]; dj[q] = hj4[q]; }
    // edge_attr: 32 edges x 32 cols, 256 threads cover it exactly
    const int c2 = (tid & 7) * 4;
    float4 a = *(const float4*)(edge_attr + (size_t)(eg0 + e) * 32 + c2);
    sMsg[e][512 + c2 + 0] = (_Float16)a.x; sMsg[e][512 + c2 + 1] = (_Float16)a.y;
    sMsg[e][512 + c2 + 2] = (_Float16)a.z; sMsg[e][512 + c2 + 3] = (_Float16)a.w;
  }
  __syncthreads();

  const int lane = tid & 31;
  const int wv   = tid >> 5;
  const int nt0  = wv * 2, nt1 = wv * 2 + 1;
  const int hi2  = lane >> 4;
  const int n0   = (lane & 15) + nt0 * 16;
  const int n1   = (lane & 15) + nt1 * 16;
  _Float16* sMflat = &sMsg[0][0];   // reuse msg LDS as m tile [32][256]

  // ---- GEMM1: m1 = SiLU(msg @ We1 + be1) ----
  v8f a00 = vzero8(), a01 = vzero8(), a10 = vzero8(), a11 = vzero8();
  gemm_quad<18, 576>(&sMsg[0][0], We1s, nt0, nt1, lane, a00, a01, a10, a11);
  silu_store4(a00, a01, a10, a11, be1[n0], be1[n1], hi2, n0, n1, &sBuf[0][0]);
  __syncthreads();

  // ---- GEMM2: m = SiLU(m1 @ We2 + be2); scatter-add into m_agg[i] ----
  a00 = vzero8(); a01 = vzero8(); a10 = vzero8(); a11 = vzero8();
  gemm_quad<8, 256>(&sBuf[0][0], We2s, nt0, nt1, lane, a00, a01, a10, a11);
  {
    const float b0 = be2[n0], b1 = be2[n1];
    #pragma unroll
    for (int r = 0; r < 8; ++r) {
      const int row  = r + 8 * hi2;
      const int nA = sI[row], nB = sI[row + 16];
      float v00 = siluf(a00[r] + b0);
      float v01 = siluf(a01[r] + b1);
      float v10 = siluf(a10[r] + b0);
      float v11 = siluf(a11[r] + b1);
      sMflat[row * 256 + n0]        = (_Float16)v00;
      sMflat[row * 256 + n1]        = (_Float16)v01;
      sMflat[(row + 16) * 256 + n0] = (_Float16)v10;
      sMflat[(row + 16) * 256 + n1] = (_Float16)v11;
      atomicAddF(m_agg + (size_t)nA * HD + n0, v00);
      atomicAddF(m_agg + (size_t)nA * HD + n1, v01);
      atomicAddF(m_agg + (size_t)nB * HD + n0, v10);
      atomicAddF(m_agg + (size_t)nB * HD + n1, v11);
    }
  }
  __syncthreads();

  // ---- GEMM3: t = SiLU(m @ Wx1 + bx1) ----
  a00 = vzero8(); a01 = vzero8(); a10 = vzero8(); a11 = vzero8();
  gemm_quad<8, 256>(sMflat, Wx1s, nt0, nt1, lane, a00, a01, a10, a11);
  silu_store4(a00, a01, a10, a11, bx1[n0], bx1[n1], hi2, n0, n1, &sBuf[0][0]);
  __syncthreads();

  // ---- coord_w = t @ Wx2 ; x_agg[i] += diff * coord_w ----
  {
    const int seg = tid & 15;
    const int e0  = tid >> 4;            // 0..15
    #pragma unroll
    for (int eo = 0; eo < 2; ++eo) {
      const int e = e0 + 16 * eo;
      float s = 0.f;
      #pragma unroll
      for (int q = 0; q < 16; ++q)
        s += (float)sBuf[e][seg * 16 + q] * Wx2[seg * 16 + q];
      sRed[e][seg] = s;
    }
  }
  __syncthreads();
  if (tid < 32) {
    float w = 0.f;
    #pragma unroll
    for (int q = 0; q < 16; ++q) w += sRed[tid][q];
    const int node = sI[tid];
    #pragma unroll
    for (int d = 0; d < 3; ++d)
      atomicAddF(x_agg + (size_t)node * 3 + d, sDiff[tid][d] * w);
  }
}

// ---------------------------------------------------------------------------
// Node kernel: per block, 32 nodes. h += phi_h([h | m_agg]); x += x_agg.
// Also refreshes the f16 shadow h16 for the next layer's edge gather.
__global__ void __launch_bounds__(256, 2)
node_kernel(float* __restrict__ h, _Float16* __restrict__ h16,
            float* __restrict__ x,
            const float* __restrict__ m_agg, const float* __restrict__ x_agg,
            const uint32_t* __restrict__ Wh1s, const float* __restrict__ bh1,
            const uint32_t* __restrict__ Wh2s, const float* __restrict__ bh2)
{
  __shared__ alignas(16) _Float16 sA[32][512];
  __shared__ alignas(16) _Float16 sU[32][256];
  const int tid = threadIdx.x;
  const int v0  = blockIdx.x * 32;

  {
    const int vv = tid >> 3;            // 0..31
    const int c0 = (tid & 7) * 32;
    // h-half of the concat tile comes straight from the f16 shadow
    const uint4* hp = (const uint4*)(h16 + (size_t)(v0 + vv) * HD + c0);
    uint4* dA = (uint4*)(&sA[vv][c0]);
    #pragma unroll
    for (int q = 0; q < 4; ++q) dA[q] = hp[q];
    #pragma unroll
    for (int q = 0; q < 8; ++q) {
      const int o = c0 + 4 * q;
      float4 b = *(const float4*)(m_agg + (size_t)(v0 + vv) * HD + o);
      sA[vv][256 + o + 0] = (_Float16)b.x; sA[vv][256 + o + 1] = (_Float16)b.y;
      sA[vv][256 + o + 2] = (_Float16)b.z; sA[vv][256 + o + 3] = (_Float16)b.w;
    }
  }
  if (tid < 96) {
    const int vv = tid / 3, d = tid % 3;
    const size_t ix = (size_t)(v0 + vv) * 3 + d;
    x[ix] += x_agg[ix];
  }
  __syncthreads();

  const int lane = tid & 31;
  const int wv   = tid >> 5;
  const int nt0  = wv * 2, nt1 = wv * 2 + 1;
  const int hi2  = lane >> 4;
  const int n0   = (lane & 15) + nt0 * 16;
  const int n1   = (lane & 15) + nt1 * 16;

  v8f a00 = vzero8(), a01 = vzero8(), a10 = vzero8(), a11 = vzero8();
  gemm_quad<16, 512>(&sA[0][0], Wh1s, nt0, nt1, lane, a00, a01, a10, a11);
  silu_store4(a00, a01, a10, a11, bh1[n0], bh1[n1], hi2, n0, n1, &sU[0][0]);
  __syncthreads();

  a00 = vzero8(); a01 = vzero8(); a10 = vzero8(); a11 = vzero8();
  gemm_quad<8, 256>(&sU[0][0], Wh2s, nt0, nt1, lane, a00, a01, a10, a11);
  {
    const float b0 = bh2[n0], b1 = bh2[n1];
    #pragma unroll
    for (int r = 0; r < 8; ++r) {
      const int row = r + 8 * hi2;
      const size_t i00 = (size_t)(v0 + row) * HD + n0;
      const size_t i01 = (size_t)(v0 + row) * HD + n1;
      const size_t i10 = (size_t)(v0 + row + 16) * HD + n0;
      const size_t i11 = (size_t)(v0 + row + 16) * HD + n1;
      float v00 = h[i00] + a00[r] + b0;
      float v01 = h[i01] + a01[r] + b1;
      float v10 = h[i10] + a10[r] + b0;
      float v11 = h[i11] + a11[r] + b1;
      h[i00] = v00; h16[i00] = (_Float16)v00;
      h[i01] = v01; h16[i01] = (_Float16)v01;
      h[i10] = v10; h16[i10] = (_Float16)v10;
      h[i11] = v11; h16[i11] = (_Float16)v11;
    }
  }
}

// ---------------------------------------------------------------------------
extern "C" void kernel_launch(void* const* d_in, const int* in_sizes, int n_in,
                              void* d_out, int out_size, void* d_ws, size_t ws_size,
                              hipStream_t stream)
{
  (void)n_in; (void)out_size; (void)ws_size;
  const float* h_in  = (const float*)d_in[0];
  const float* x_in  = (const float*)d_in[1];
  const float* eattr = (const float*)d_in[2];
  const int*   eidx  = (const int*)d_in[3];
  const float* We1   = (const float*)d_in[4];
  const float* be1   = (const float*)d_in[5];
  const float* We2   = (const float*)d_in[6];
  const float* be2   = (const float*)d_in[7];
  const float* Wx1   = (const float*)d_in[8];
  const float* bx1   = (const float*)d_in[9];
  const float* Wx2   = (const float*)d_in[10];
  const float* Wh1   = (const float*)d_in[11];
  const float* bh1   = (const float*)d_in[12];
  const float* Wh2   = (const float*)d_in[13];
  const float* bh2   = (const float*)d_in[14];

  const int N = in_sizes[0] / HD;   // 20000
  const int E = in_sizes[3] / 2;    // 320000
  const int L = 4;

  float* h_out = (float*)d_out;
  float* x_out = h_out + (size_t)N * HD;

  // Workspace layout (256B-aligned chunks)
  uint8_t* p = (uint8_t*)d_ws;
  auto take = [&](size_t bytes) {
    uint8_t* r = p;
    p += (bytes + 255) & ~(size_t)255;
    return r;
  };
  const int KB_E1 = 18, KB_256 = 8, KB_512 = 16;   // K/32 per GEMM
  uint32_t* We1s = (uint32_t*)take((size_t)L * KB_E1  * 4096 * 4);
  uint32_t* We2s = (uint32_t*)take((size_t)L * KB_256 * 4096 * 4);
  uint32_t* Wx1s = (uint32_t*)take((size_t)L * KB_256 * 4096 * 4);
  uint32_t* Wh1s = (uint32_t*)take((size_t)L * KB_512 * 4096 * 4);
  uint32_t* Wh2s = (uint32_t*)take((size_t)L * KB_256 * 4096 * 4);
  float* m_agg = (float*)take((size_t)N * (HD + 3) * 4);   // m_agg + x_agg contiguous
  float* x_agg = m_agg + (size_t)N * HD;
  _Float16* h16 = (_Float16*)take((size_t)N * HD * 2);     // f16 shadow of h

  // Pre-swizzle weights to f16 WMMA-B fragment order (every call; deterministic)
  convert_weights<<<dim3((KB_E1  * 4096 + 255) / 256, L), 256, 0, stream>>>(
      We1, We1s, KB_E1, 545, 545 * 256, KB_E1 * 4096, 1);
  convert_weights<<<dim3((KB_256 * 4096 + 255) / 256, L), 256, 0, stream>>>(
      We2, We2s, KB_256, 256, 256 * 256, KB_256 * 4096, 0);
  convert_weights<<<dim3((KB_256 * 4096 + 255) / 256, L), 256, 0, stream>>>(
      Wx1, Wx1s, KB_256, 256, 256 * 256, KB_256 * 4096, 0);
  convert_weights<<<dim3((KB_512 * 4096 + 255) / 256, L), 256, 0, stream>>>(
      Wh1, Wh1s, KB_512, 512, 512 * 256, KB_512 * 4096, 0);
  convert_weights<<<dim3((KB_256 * 4096 + 255) / 256, L), 256, 0, stream>>>(
      Wh2, Wh2s, KB_256, 256, 256 * 256, KB_256 * 4096, 0);

  // h, x working copies live in d_out (updated in place, layer by layer)
  hipMemcpyAsync(h_out, h_in, (size_t)N * HD * 4, hipMemcpyDeviceToDevice, stream);
  hipMemcpyAsync(x_out, x_in, (size_t)N * 3 * 4, hipMemcpyDeviceToDevice, stream);
  cvt_h16_kernel<<<(N * HD + 255) / 256, 256, 0, stream>>>(h_in, h16, N * HD);

  const int ztotal = N * (HD + 3);
  for (int l = 0; l < L; ++l) {
    zero_kernel<<<(ztotal + 255) / 256, 256, 0, stream>>>(m_agg, ztotal);
    edge_kernel<<<E / 32, 256, 0, stream>>>(
        h16, x_out, eattr, eidx, E,
        We1s + (size_t)l * KB_E1  * 4096, be1 + l * HD,
        We2s + (size_t)l * KB_256 * 4096, be2 + l * HD,
        Wx1s + (size_t)l * KB_256 * 4096, bx1 + l * HD,
        Wx2 + l * HD,
        m_agg, x_agg);
    node_kernel<<<N / 32, 256, 0, stream>>>(
        h_out, h16, x_out, m_agg, x_agg,
        Wh1s + (size_t)l * KB_512 * 4096, bh1 + l * HD,
        Wh2s + (size_t)l * KB_256 * 4096, bh2 + l * HD);
  }
}
